// MemoryAttentionNetwork_32693291057188
// MI455X (gfx1250) — compile-verified, matrix operand order checked by reading, not run on hardware
//
#include <hip/hip_runtime.h>
#include <hip/hip_bf16.h>

// ---------------------------------------------------------------------------
// MI455X (gfx1250) implementation. wave32, WMMA bf16 16x16x32, fp32 accum.
// Tile loads on the hot GEMM path use GLOBAL_LOAD_ASYNC_TO_LDS_B128 (ASYNCcnt).
// ---------------------------------------------------------------------------

typedef __bf16 bf16;
typedef __attribute__((ext_vector_type(16))) __bf16 v16bf;
typedef __attribute__((ext_vector_type(8)))  __bf16 bf16x8;
typedef __attribute__((ext_vector_type(8)))  float  v8f;

#define DIMC   1024
#define BB     8
#define LMEMN  256
#define AHN    6208          // 4*(1024+512) + 64
#define KVN    6464          // 256 + AHN
#define HNUM   8
#define DH     128
#define LDT    40            // LDS tile row stride (32 data halves + 8 pad)

// ---- fragment load from LDS (documented 16-bit A/B 16x32 lane layout) -----
// lane L (0..31): row = frag_row + (L%16); halves j=0..7 -> k = klo..klo+7,
// j=8..15 -> k = klo+16..klo+23, klo = (L>=16)?8:0. Two 16B LDS reads.
__device__ __forceinline__ v16bf load_frag(const bf16* tile, int row, int lane) {
  const bf16* p = tile + row * LDT + ((lane & 16) ? 8 : 0);
  bf16x8 lo = *(const bf16x8*)p;
  bf16x8 hi = *(const bf16x8*)(p + 16);
  return __builtin_shufflevector(lo, hi, 0,1,2,3,4,5,6,7,8,9,10,11,12,13,14,15);
}

// 8 WMMAs covering this wave's 32x64 sub-tile of the 128x128 block tile.
__device__ __forceinline__ void mma_tiles(const bf16* As, const bf16* Bs,
                                          int m0, int n0, int lane, v8f acc[2][4]) {
  v16bf af0 = load_frag(As, m0 +      (lane & 15), lane);
  v16bf af1 = load_frag(As, m0 + 16 + (lane & 15), lane);
  #pragma unroll
  for (int ni = 0; ni < 4; ni++) {
    v16bf bfr = load_frag(Bs, n0 + ni * 16 + (lane & 15), lane);
    acc[0][ni] = __builtin_amdgcn_wmma_f32_16x16x32_bf16(
        false, af0, false, bfr, (short)0, acc[0][ni], false, false);
    acc[1][ni] = __builtin_amdgcn_wmma_f32_16x16x32_bf16(
        false, af1, false, bfr, (short)0, acc[1][ni], false, false);
  }
}

// ---------------------------------------------------------------------------
// Generic batched GEMM: C[M,N] (+)= A[M,K](bf16,row-major) * Bt[N,K](bf16).
// EPI: 0 = store fp32, 1 = store bf16, 2 = accumulate into fp32.
// 2D batch: offs = (z/nh)*stride + (z%nh)*stride_h.
// ---------------------------------------------------------------------------
template<int EPI>
__global__ __launch_bounds__(256) void gemm_bf16_wmma(
    const bf16* __restrict__ Abase, const bf16* __restrict__ Btbase,
    float* __restrict__ Cf, bf16* __restrict__ Cb,
    int M, int N, int K, int lda, int ldb, int ldc,
    long long sA, long long sB, long long sC,
    int nh, long long sAh, long long sCh)
{
  __shared__ __align__(16) bf16 As[128 * LDT];
  __shared__ __align__(16) bf16 Bs[128 * LDT];

  const int z = blockIdx.z;
  const bf16* A  = Abase  + (long long)(z / nh) * sA + (long long)(z % nh) * sAh;
  const bf16* Bt = Btbase + (long long)z * sB;
  const long long coff = (long long)(z / nh) * sC + (long long)(z % nh) * sCh;

  const int tileM = blockIdx.y * 128, tileN = blockIdx.x * 128;
  const int t = threadIdx.x, lane = t & 31, w = t >> 5;
  const int m0 = (w >> 1) * 32, n0 = (w & 1) * 64;
  const int rowL = t >> 1, cH = (t & 1) * 16;

  const v8f vz = {0.f,0.f,0.f,0.f,0.f,0.f,0.f,0.f};
  v8f acc[2][4];
  #pragma unroll
  for (int i = 0; i < 2; i++)
    #pragma unroll
    for (int j = 0; j < 4; j++) acc[i][j] = vz;

  const int gA = tileM + rowL;      // global A row for this thread's load slot
  const int gB = tileN + rowL;      // global Bt row (N assumed multiple of 128)

  if (tileM + 128 <= M) {
    // ---- fast path: whole tile in range -> async copy global->LDS --------
    // Each lane moves 2x16B for A and 2x16B for B per K-chunk; ASYNCcnt
    // tracks completion; instruction "offset:" applies to BOTH addresses.
    const unsigned ldsA = (unsigned)(unsigned long long)&As[rowL * LDT + cH];
    const unsigned ldsB = (unsigned)(unsigned long long)&Bs[rowL * LDT + cH];
    const unsigned long long aBase = (unsigned long long)A;
    const unsigned long long bBase = (unsigned long long)Bt;
    const long long arow = (long long)gA * lda + cH;
    const long long brow = (long long)gB * ldb + cH;
    for (int k0 = 0; k0 < K; k0 += 32) {
      const unsigned offA = (unsigned)((arow + k0) * 2);
      const unsigned offB = (unsigned)((brow + k0) * 2);
      __syncthreads();    // previous chunk fully consumed before overwrite
      asm volatile(
        "global_load_async_to_lds_b128 %0, %2, %4\n\t"
        "global_load_async_to_lds_b128 %0, %2, %4 offset:16\n\t"
        "global_load_async_to_lds_b128 %1, %3, %5\n\t"
        "global_load_async_to_lds_b128 %1, %3, %5 offset:16\n\t"
        "s_wait_asynccnt 0x0"
        :: "v"(ldsA), "v"(ldsB), "v"(offA), "v"(offB), "s"(aBase), "s"(bBase)
        : "memory");
      __syncthreads();
      mma_tiles(As, Bs, m0, n0, lane, acc);
    }
  } else {
    // ---- ragged last M-tile: guarded register-staged path ----------------
    const bf16 z0 = (bf16)0.f;
    const bf16x8 ZB8 = {z0,z0,z0,z0,z0,z0,z0,z0};
    for (int k0 = 0; k0 < K; k0 += 32) {
      bf16x8 a0 = ZB8, a1 = ZB8;
      if (gA < M) {
        const bf16* pa = A + (long long)gA * lda + k0 + cH;
        a0 = *(const bf16x8*)pa;
        a1 = *(const bf16x8*)(pa + 8);
        if (k0 + 32 < K) __builtin_prefetch((const void*)(pa + 32), 0, 1);
      }
      const bf16* pb = Bt + (long long)gB * ldb + k0 + cH;
      bf16x8 b0 = *(const bf16x8*)pb;
      bf16x8 b1 = *(const bf16x8*)(pb + 8);

      __syncthreads();
      *(bf16x8*)&As[rowL * LDT + cH]     = a0;
      *(bf16x8*)&As[rowL * LDT + cH + 8] = a1;
      *(bf16x8*)&Bs[rowL * LDT + cH]     = b0;
      *(bf16x8*)&Bs[rowL * LDT + cH + 8] = b1;
      __syncthreads();
      mma_tiles(As, Bs, m0, n0, lane, acc);
    }
  }

  // C/D layout: VGPR i, lanes 0-15 -> M=i, lanes 16-31 -> M=8+i; N = lane%16.
  const int rb = tileM + m0 + ((lane >> 4) << 3);
  const int cb = tileN + n0 + (lane & 15);
  #pragma unroll
  for (int mi = 0; mi < 2; mi++)
    #pragma unroll
    for (int ni = 0; ni < 4; ni++)
      #pragma unroll
      for (int i = 0; i < 8; i++) {
        int r = rb + mi * 16 + i;
        if (r < M) {
          long long idx = coff + (long long)r * ldc + cb + ni * 16;
          float v = acc[mi][ni][i];
          if (EPI == 0)      Cf[idx] = v;
          else if (EPI == 1) Cb[idx] = (bf16)v;
          else               Cf[idx] += v;
        }
      }
}

// ---------------------------------------------------------------------------
// ctx[b,h,d,e] = krcp[d] * sum_n exp(k[n,d]-kmax[d]) * v[n,e]
// kv: (B, KVN, 2048) bf16 (cols 0..1023 = k, 1024..2047 = v).
// Output transposed bf16: ctxT[(b*8+h)*128 + e][d]  (Bt layout for q@ctx).
// ---------------------------------------------------------------------------
__global__ __launch_bounds__(256) void ctx_wmma_kernel(
    const bf16* __restrict__ kv, const float* __restrict__ kmax,
    const float* __restrict__ krcp, bf16* __restrict__ ctxT)
{
  __shared__ __align__(16) bf16 Ks[128 * LDT];   // rows d, cols n-local
  __shared__ __align__(16) bf16 Vs[128 * LDT];   // rows e, cols n-local
  const int z = blockIdx.x, b = z >> 3, h = z & 7;
  const int t = threadIdx.x, lane = t & 31, w = t >> 5;
  const int d = t & 127, ng = (t >> 7) * 16;
  const float km = kmax[b * DIMC + h * DH + d];
  const bf16* kb = kv + (long long)b * KVN * 2048 +        h * DH + d;
  const bf16* vb = kv + (long long)b * KVN * 2048 + 1024 + h * DH + d;
  const int m0 = (w >> 1) * 32, n0 = (w & 1) * 64;

  const v8f vz = {0.f,0.f,0.f,0.f,0.f,0.f,0.f,0.f};
  v8f acc[2][4];
  #pragma unroll
  for (int i = 0; i < 2; i++)
    #pragma unroll
    for (int j = 0; j < 4; j++) acc[i][j] = vz;

  for (int nb = 0; nb < KVN; nb += 32) {
    alignas(16) bf16 kl[16];
    alignas(16) bf16 vl[16];
    #pragma unroll
    for (int i = 0; i < 16; i++) {
      long long ro = (long long)(nb + ng + i) * 2048;
      kl[i] = (bf16)__expf((float)kb[ro] - km);
      vl[i] = vb[ro];
    }
    __syncthreads();
    *(bf16x8*)&Ks[d * LDT + ng]     = *(const bf16x8*)kl;
    *(bf16x8*)&Ks[d * LDT + ng + 8] = *(const bf16x8*)(kl + 8);
    *(bf16x8*)&Vs[d * LDT + ng]     = *(const bf16x8*)vl;
    *(bf16x8*)&Vs[d * LDT + ng + 8] = *(const bf16x8*)(vl + 8);
    __syncthreads();
    mma_tiles(Ks, Vs, m0, n0, lane, acc);
  }

  const int rb = m0 + ((lane >> 4) << 3);
  const int cb = n0 + (lane & 15);
  #pragma unroll
  for (int mi = 0; mi < 2; mi++)
    #pragma unroll
    for (int ni = 0; ni < 4; ni++) {
      int ee = cb + ni * 16;
      #pragma unroll
      for (int i = 0; i < 8; i++) {
        int dd = rb + mi * 16 + i;
        float rs = krcp[b * DIMC + h * DH + dd];   // precomputed 1/sumexp
        ctxT[((long long)z * DH + ee) * DH + dd] = (bf16)(acc[mi][ni][i] * rs);
      }
    }
}

// ---------------------------------------------------------------------------
// block (256 threads) reduction of two sums, via wave32 shuffles + LDS
// ---------------------------------------------------------------------------
__device__ __forceinline__ void block_reduce2(float& s1, float& s2, float* sh) {
  #pragma unroll
  for (int o = 16; o; o >>= 1) { s1 += __shfl_xor(s1, o, 32); s2 += __shfl_xor(s2, o, 32); }
  const int lane = threadIdx.x & 31, w = threadIdx.x >> 5;
  if (lane == 0) { sh[w] = s1; sh[8 + w] = s2; }
  __syncthreads();
  float a = 0.f, b = 0.f;
  #pragma unroll
  for (int i = 0; i < 8; i++) { a += sh[i]; b += sh[8 + i]; }
  s1 = a; s2 = b;
  __syncthreads();
}

// nx = LN(x) (no affine), bf16 out.  One block per 1024-wide row.
__global__ __launch_bounds__(256) void ln_rows_bf16(
    const float* __restrict__ in, bf16* __restrict__ out)
{
  __shared__ float sh[16];
  const int r = blockIdx.x, t = threadIdx.x;
  const float* p = in + (long long)r * DIMC;
  float x[4], s = 0.f, s2 = 0.f;
  #pragma unroll
  for (int i = 0; i < 4; i++) { x[i] = p[t + i * 256]; s += x[i]; s2 += x[i] * x[i]; }
  block_reduce2(s, s2, sh);
  float mean = s * (1.f / 1024.f);
  float rstd = rsqrtf(s2 * (1.f / 1024.f) - mean * mean + 1e-5f);
  bf16* q = out + (long long)r * DIMC;
  #pragma unroll
  for (int i = 0; i < 4; i++) q[t + i * 256] = (bf16)((x[i] - mean) * rstd);
}

// out = LN(in)*g + b, fp32 out (for out_lmem).
__global__ __launch_bounds__(256) void ln_affine_kernel(
    const float* __restrict__ in, const float* __restrict__ g,
    const float* __restrict__ bb, float* __restrict__ out)
{
  __shared__ float sh[16];
  const int r = blockIdx.x, t = threadIdx.x;
  const float* p = in + (long long)r * DIMC;
  float x[4], s = 0.f, s2 = 0.f;
  #pragma unroll
  for (int i = 0; i < 4; i++) { x[i] = p[t + i * 256]; s += x[i]; s2 += x[i] * x[i]; }
  block_reduce2(s, s2, sh);
  float mean = s * (1.f / 1024.f);
  float rstd = rsqrtf(s2 * (1.f / 1024.f) - mean * mean + 1e-5f);
  float* q = out + (long long)r * DIMC;
  #pragma unroll
  for (int i = 0; i < 4; i++) {
    int c = t + i * 256;
    q[c] = (x[i] - mean) * rstd * g[c] + bb[c];
  }
}

// out_mem = LN(queue[:, :, -1024:, :]) * g + b  where queue = cat(smem,hiddens).
// row r in (4,8,1024): i<512 -> smem[d][b][512+i], else hiddens[d][b][i-512].
__global__ __launch_bounds__(256) void out_mem_kernel(
    const float* __restrict__ smem, const float* __restrict__ hiddens,
    const float* __restrict__ g, const float* __restrict__ bb,
    float* __restrict__ out)
{
  __shared__ float sh[16];
  const int r = blockIdx.x, t = threadIdx.x;
  const int dep = r >> 13, rem = r & 8191, b = rem >> 10, i0 = rem & 1023;
  const float* p = (i0 < 512)
      ? smem    + ((((long long)dep * 8 + b) * 1024) + 512 + i0) * DIMC
      : hiddens + ((((long long)dep * 8 + b) * 512) + (i0 - 512)) * DIMC;
  float x[4], s = 0.f, s2 = 0.f;
  #pragma unroll
  for (int i = 0; i < 4; i++) { x[i] = p[t + i * 256]; s += x[i]; s2 += x[i] * x[i]; }
  block_reduce2(s, s2, sh);
  float mean = s * (1.f / 1024.f);
  float rstd = rsqrtf(s2 * (1.f / 1024.f) - mean * mean + 1e-5f);
  float* q = out + (long long)r * DIMC;
  #pragma unroll
  for (int i = 0; i < 4; i++) {
    int c = t + i * 256;
    q[c] = (x[i] - mean) * rstd * g[c] + bb[c];
  }
}

// q softmax over head dim (128). One 128-thread block per (b,n,h) row.
__global__ __launch_bounds__(128) void qsoftmax_kernel(
    const float* __restrict__ q, bf16* __restrict__ qs)
{
  __shared__ float sh[4];
  const int r = blockIdx.x;
  const long long bn = r >> 3; const int h = r & 7;
  const float* src = q + bn * DIMC + h * DH;
  const int t = threadIdx.x, lane = t & 31, w = t >> 5;
  float v = src[t];
  float m = v;
  #pragma unroll
  for (int o = 16; o; o >>= 1) m = fmaxf(m, __shfl_xor(m, o, 32));
  if (lane == 0) sh[w] = m;
  __syncthreads();
  m = fmaxf(fmaxf(sh[0], sh[1]), fmaxf(sh[2], sh[3]));
  float e = __expf(v - m);
  float s = e;
  #pragma unroll
  for (int o = 16; o; o >>= 1) s += __shfl_xor(s, o, 32);
  __syncthreads();
  if (lane == 0) sh[w] = s;
  __syncthreads();
  s = sh[0] + sh[1] + sh[2] + sh[3];
  qs[bn * DIMC + h * DH + t] = (bf16)(e * __builtin_amdgcn_rcpf(s));
}

// Online softmax stats for k columns: per (b, col<1024) over N=6464 rows.
// Stores max and RECIPROCAL of sum-of-exp (so consumers only multiply).
__global__ __launch_bounds__(256) void kstats_kernel(
    const bf16* __restrict__ kv, float* __restrict__ kmax, float* __restrict__ krcp)
{
  const int col = blockIdx.x * 256 + threadIdx.x;   // grid (4, 8)
  const int b = blockIdx.y;
  const bf16* p = kv + (long long)b * KVN * 2048 + col;
  float m = -1e30f, s = 0.f;
  for (int n = 0; n < KVN; n++) {
    float v = (float)p[(long long)n * 2048];
    if (v > m) { s = s * __expf(m - v) + 1.f; m = v; }
    else       { s += __expf(v - m); }
  }
  kmax[b * DIMC + col] = m;
  krcp[b * DIMC + col] = 1.f / s;
}

// Gate: a=1+tanh(Ga+Ua_b+Wa_b); c=sig(Gc+Uc_b+Wc_b); g=tanh(Gu+U_b+a*h)
// h_new = c*h + (1-c)*g; write fp32 + bf16.
__global__ __launch_bounds__(256) void gate_kernel(
    const float* __restrict__ Ga, const float* __restrict__ Gc,
    const float* __restrict__ Gu,
    const float* __restrict__ Uab, const float* __restrict__ Wab,
    const float* __restrict__ Ucb, const float* __restrict__ Wcb,
    const float* __restrict__ Ub,
    float* __restrict__ h, bf16* __restrict__ hb)
{
  const long long i = (long long)blockIdx.x * 256 + threadIdx.x;
  const int c = (int)(i & 1023);
  float hv = h[i];
  float a  = 1.f + tanhf(Ga[i] + Uab[c] + Wab[c]);
  float cg = __builtin_amdgcn_rcpf(1.f + __expf(-(Gc[i] + Ucb[c] + Wcb[c])));
  float gg = tanhf(Gu[i] + Ub[c] + a * hv);
  float nh = cg * hv + (1.f - cg) * gg;
  h[i] = nh;
  hb[i] = (bf16)nh;
}

// next_lmem0 = lmem + lmem_pos_emb (broadcast over batch); fp32 + bf16.
__global__ __launch_bounds__(256) void init_lmem_kernel(
    const float* __restrict__ lmem, const float* __restrict__ pe,
    float* __restrict__ h, bf16* __restrict__ hb)
{
  const long long i = (long long)blockIdx.x * 256 + threadIdx.x;
  float v = lmem[i] + pe[i % (LMEMN * DIMC)];
  h[i] = v;
  hb[i] = (bf16)v;
}

// ah[b, r, c]  (B, 6208, 1024) bf16: cat(smem,hiddens)+depth_emb, then mem_kv.
__global__ __launch_bounds__(256) void build_ah_kernel(
    const float* __restrict__ smem, const float* __restrict__ hiddens,
    const float* __restrict__ demb, const float* __restrict__ memkv,
    bf16* __restrict__ ah)
{
  const long long i = (long long)blockIdx.x * 256 + threadIdx.x;
  const int c = (int)(i & 1023);
  const long long rb = i >> 10;
  const int r = (int)(rb % AHN);
  const long long b = rb / AHN;
  float v;
  if (r < 6144) {
    const int dep = r / 1536, j = r % 1536;
    if (j < 1024) v = smem[(((long long)dep * 8 + b) * 1024 + j) * DIMC + c];
    else          v = hiddens[(((long long)dep * 8 + b) * 512 + (j - 1024)) * DIMC + c];
    v += demb[dep];
  } else {
    v = memkv[(long long)(r - 6144) * DIMC + c];
  }
  ah[i] = (bf16)v;
}

__global__ __launch_bounds__(256) void conv_bf16_kernel(
    const float* __restrict__ src, bf16* __restrict__ dst)
{
  const long long i = (long long)blockIdx.x * 256 + threadIdx.x;
  dst[i] = (bf16)src[i];
}

// dst[n*K + k] = src[k*N + n]  (make Bt(N,K) from row-major (K,N) weight)
__global__ __launch_bounds__(256) void conv_bf16_T_kernel(
    const float* __restrict__ src, bf16* __restrict__ dst, int K, int N)
{
  const long long i = (long long)blockIdx.x * 256 + threadIdx.x;
  const int n = (int)(i / K), k = (int)(i % K);
  dst[i] = (bf16)src[(long long)k * N + n];
}

// ---------------------------------------------------------------------------
static void launch_gemm(int epi, const bf16* A, const bf16* Bt, float* Cf, bf16* Cb,
    int M, int N, int K, int lda, int ldb, int ldc,
    long long sA, long long sB, long long sC,
    int nh, long long sAh, long long sCh, int batch, hipStream_t st)
{
  dim3 g(N / 128, (M + 127) / 128, batch);
  if (epi == 0)
    gemm_bf16_wmma<0><<<g, 256, 0, st>>>(A, Bt, Cf, Cb, M, N, K, lda, ldb, ldc, sA, sB, sC, nh, sAh, sCh);
  else if (epi == 1)
    gemm_bf16_wmma<1><<<g, 256, 0, st>>>(A, Bt, Cf, Cb, M, N, K, lda, ldb, ldc, sA, sB, sC, nh, sAh, sCh);
  else
    gemm_bf16_wmma<2><<<g, 256, 0, st>>>(A, Bt, Cf, Cb, M, N, K, lda, ldb, ldc, sA, sB, sC, nh, sAh, sCh);
}

extern "C" void kernel_launch(void* const* d_in, const int* in_sizes, int n_in,
                              void* d_out, int out_size, void* d_ws, size_t ws_size,
                              hipStream_t stream) {
  (void)in_sizes; (void)n_in; (void)out_size; (void)ws_size;
  const float* lmem     = (const float*)d_in[0];
  const float* smem     = (const float*)d_in[1];
  const float* hiddens  = (const float*)d_in[2];
  const float* w_q      = (const float*)d_in[3];
  const float* w_kv     = (const float*)d_in[4];
  const float* w_out    = (const float*)d_in[5];
  const float* depth_e  = (const float*)d_in[6];
  const float* lmem_pe  = (const float*)d_in[7];
  const float* mem_kv   = (const float*)d_in[8];
  const float* Ua_w = (const float*)d_in[9];  const float* Ua_b = (const float*)d_in[10];
  const float* Wa_w = (const float*)d_in[11]; const float* Wa_b = (const float*)d_in[12];
  const float* Uc_w = (const float*)d_in[13]; const float* Uc_b = (const float*)d_in[14];
  const float* Wc_w = (const float*)d_in[15]; const float* Wc_b = (const float*)d_in[16];
  const float* U_w  = (const float*)d_in[17]; const float* U_b  = (const float*)d_in[18];
  const float* n1g = (const float*)d_in[19]; const float* n1b = (const float*)d_in[20];
  const float* n2g = (const float*)d_in[21]; const float* n2b = (const float*)d_in[22];
  float* outp = (float*)d_out;

  // ---- workspace bump allocator ----
  size_t off = 0;
  char* base = (char*)d_ws;
  auto alloc = [&](size_t bytes) -> char* {
    char* p = base + off;
    off += (bytes + 255) & ~(size_t)255;
    return p;
  };
  bf16*  ah16  = (bf16*)alloc((size_t)BB * AHN * DIMC * 2);          // 101.7 MB
  bf16*  kvF   = (bf16*)alloc((size_t)BB * KVN * 2048 * 2);          // 202 MB
  bf16*  wqT   = (bf16*)alloc((size_t)DIMC * DIMC * 2);
  bf16*  wkvT  = (bf16*)alloc((size_t)2048 * DIMC * 2);
  bf16*  woutT = (bf16*)alloc((size_t)DIMC * DIMC * 2);
  bf16*  ua16  = (bf16*)alloc((size_t)DIMC * DIMC * 2);
  bf16*  wa16  = (bf16*)alloc((size_t)DIMC * DIMC * 2);
  bf16*  uc16  = (bf16*)alloc((size_t)DIMC * DIMC * 2);
  bf16*  wc16  = (bf16*)alloc((size_t)DIMC * DIMC * 2);
  bf16*  uu16  = (bf16*)alloc((size_t)DIMC * DIMC * 2);
  float* h32   = (float*)alloc((size_t)BB * LMEMN * DIMC * 4);
  bf16*  h16   = (bf16*)alloc((size_t)BB * LMEMN * DIMC * 2);
  bf16*  nx16  = (bf16*)alloc((size_t)BB * LMEMN * DIMC * 2);
  float* q32   = (float*)alloc((size_t)BB * LMEMN * DIMC * 4);
  bf16*  qs16  = (bf16*)alloc((size_t)BB * LMEMN * DIMC * 2);
  float* kmaxB = (float*)alloc((size_t)BB * DIMC * 4);
  float* krcpB = (float*)alloc((size_t)BB * DIMC * 4);
  bf16*  ctxT  = (bf16*)alloc((size_t)BB * HNUM * DH * DH * 2);
  bf16*  ho16  = (bf16*)alloc((size_t)BB * LMEMN * DIMC * 2);
  bf16*  ax16  = (bf16*)alloc((size_t)BB * LMEMN * DIMC * 2);
  float* Ga    = (float*)alloc((size_t)BB * LMEMN * DIMC * 4);
  float* Gc    = (float*)alloc((size_t)BB * LMEMN * DIMC * 4);
  float* Gu    = (float*)alloc((size_t)BB * LMEMN * DIMC * 4);

  // ---- one-time preparation ----
  conv_bf16_T_kernel<<<4096, 256, 0, stream>>>(w_q,   wqT,   DIMC, DIMC);
  conv_bf16_T_kernel<<<8192, 256, 0, stream>>>(w_kv,  wkvT,  DIMC, 2048);
  conv_bf16_T_kernel<<<4096, 256, 0, stream>>>(w_out, woutT, DIMC, DIMC);
  conv_bf16_kernel<<<4096, 256, 0, stream>>>(Ua_w, ua16);   // t@W.T => Bt = W as-is
  conv_bf16_kernel<<<4096, 256, 0, stream>>>(Wa_w, wa16);
  conv_bf16_kernel<<<4096, 256, 0, stream>>>(Uc_w, uc16);
  conv_bf16_kernel<<<4096, 256, 0, stream>>>(Wc_w, wc16);
  conv_bf16_kernel<<<4096, 256, 0, stream>>>(U_w,  uu16);
  init_lmem_kernel<<<8192, 256, 0, stream>>>(lmem, lmem_pe, h32, h16);
  build_ah_kernel<<<(BB * AHN * DIMC) / 256, 256, 0, stream>>>(smem, hiddens, depth_e, mem_kv, ah16);

  // kv of the ah rows: iteration-invariant, computed ONCE. (dominant GEMM)
  launch_gemm(1, ah16, wkvT, nullptr, kvF + (long long)LMEMN * 2048,
              AHN, 2048, DIMC, DIMC, DIMC, 2048,
              (long long)AHN * DIMC, 0, (long long)KVN * 2048, 1, 0, 0, BB, stream);

  const long long sH = (long long)LMEMN * DIMC;   // per-batch lmem row stride
  for (int it = 0; it < 2; it++) {
    ln_rows_bf16<<<BB * LMEMN, 256, 0, stream>>>(h32, nx16);
    // q = nx @ w_q
    launch_gemm(0, nx16, wqT, q32, nullptr, BB * LMEMN, DIMC, DIMC,
                DIMC, DIMC, DIMC, 0, 0, 0, 1, 0, 0, 1, stream);
    // kv rows [0,256) per batch = nx @ w_kv
    launch_gemm(1, nx16, wkvT, nullptr, kvF, LMEMN, 2048, DIMC,
                DIMC, DIMC, 2048, sH, 0, (long long)KVN * 2048, 1, 0, 0, BB, stream);
    qsoftmax_kernel<<<BB * LMEMN * HNUM, 128, 0, stream>>>(q32, qs16);
    kstats_kernel<<<dim3(4, BB), 256, 0, stream>>>(kvF, kmaxB, krcpB);
    ctx_wmma_kernel<<<BB * HNUM, 256, 0, stream>>>(kvF, kmaxB, krcpB, ctxT);
    // out[b,n,h,e] = q_soft[b,n,h,:] @ ctx[b,h,:,e]   (batch = b*8+h)
    launch_gemm(1, qs16, ctxT, nullptr, ho16, LMEMN, DH, DH,
                DIMC, DH, DIMC, sH, (long long)DH * DH, sH,
                HNUM, DH, DH, BB * HNUM, stream);
    // attn_out (bf16 only; consumed by gate linears) = headout @ w_out
    launch_gemm(1, ho16, woutT, nullptr, ax16, BB * LMEMN, DIMC, DIMC,
                DIMC, DIMC, DIMC, 0, 0, 0, 1, 0, 0, 1, stream);
    // gate linears
    launch_gemm(0, ax16, ua16, Ga, nullptr, BB * LMEMN, DIMC, DIMC, DIMC, DIMC, DIMC, 0,0,0,1,0,0,1, stream);
    launch_gemm(2, h16,  wa16, Ga, nullptr, BB * LMEMN, DIMC, DIMC, DIMC, DIMC, DIMC, 0,0,0,1,0,0,1, stream);
    launch_gemm(0, ax16, uc16, Gc, nullptr, BB * LMEMN, DIMC, DIMC, DIMC, DIMC, DIMC, 0,0,0,1,0,0,1, stream);
    launch_gemm(2, h16,  wc16, Gc, nullptr, BB * LMEMN, DIMC, DIMC, DIMC, DIMC, DIMC, 0,0,0,1,0,0,1, stream);
    launch_gemm(0, ax16, uu16, Gu, nullptr, BB * LMEMN, DIMC, DIMC, DIMC, DIMC, DIMC, 0,0,0,1,0,0,1, stream);
    gate_kernel<<<8192, 256, 0, stream>>>(Ga, Gc, Gu, Ua_b, Wa_b, Uc_b, Wc_b, U_b, h32, h16);
  }

  // ---- outputs ----
  out_mem_kernel<<<4 * 8 * 1024, 256, 0, stream>>>(smem, hiddens, n1g, n1b, outp);
  ln_affine_kernel<<<BB * LMEMN, 256, 0, stream>>>(h32, n2g, n2b, outp + 33554432LL);
}
